// MyModel_87522843561154
// MI455X (gfx1250) — matrix operand order repaired and checked by reading, not compiled, not run on hardware
//
#include <hip/hip_runtime.h>

// ---------------------------------------------------------------------------
// CDNA5 / gfx1250 nested-LSTM kernel.
//  - wave32, one 16-row batch tile per wave
//  - h@U recurrent GEMMs on v_wmma_f32_16x16x32_f16 (K=32 == U1)
//  - D-layout -> A-layout recurrence via per-wave LDS roundtrip
// ---------------------------------------------------------------------------

typedef __attribute__((ext_vector_type(16))) _Float16 v16h;
typedef __attribute__((ext_vector_type(8)))  _Float16 v8h;
typedef __attribute__((ext_vector_type(8)))  float    v8f;

// Intra-wave LDS store->load ordering fence (CDNA5 split counters).
#define LDS_FENCE() asm volatile("s_wait_dscnt 0x0" ::: "memory")

__device__ __forceinline__ float fast_sigmoid(float x) {
  return __builtin_amdgcn_rcpf(1.0f + __expf(-x));
}
__device__ __forceinline__ float fast_tanh(float x) {
  // tanh(x) = 1 - 2*e / (1 + e), e = exp(-2x)
  float e = __expf(-2.0f * x);
  return fmaf(-2.0f * e, __builtin_amdgcn_rcpf(1.0f + e), 1.0f);
}

// ===========================================================================
// Inner LSTM: 65536 independent sequences, S=20 steps, D=1, U1=32.
// One wave handles 16 rows. Per step: 8x v_wmma_f32_16x16x32_f16.
// Writes final hidden state (features) as fp16 [rows][32] to workspace.
// ===========================================================================
__global__ __launch_bounds__(256, 1) void lstm_inner_kernel(
    const float* __restrict__ x,    // [B*T, 20, 1]
    const float* __restrict__ Wi,   // [1, 128]
    const float* __restrict__ Ui,   // [32, 128]
    const float* __restrict__ bi,   // [128]
    _Float16* __restrict__ feats)   // [B*T, 32] fp16 out
{
  constexpr int S = 20;
  constexpr int G = 128;  // 4*U1

  __shared__ __attribute__((aligned(16))) _Float16 hbuf[8][16 * 32]; // per-wave h tile
  __shared__ __attribute__((aligned(16))) float    xls[8][16 * S];   // per-wave x tile

  const int lane = threadIdx.x & 31;
  const int wv   = threadIdx.x >> 5;
  const int n0   = (blockIdx.x * 8 + wv) * 16;   // first sequence row of this wave
  const int hi   = lane >> 4;                    // lane half (A/B layout K split)
  const int l16  = lane & 15;
  const int klo  = hi * 8;

  // ---- Ui in WMMA B layout (8 column tiles); Wi row + bias per column ----
  v16h bU[8];
  float wiv[8], bzv[8];
#pragma unroll
  for (int t = 0; t < 8; ++t) {
    const int col = t * 16 + l16;
#pragma unroll
    for (int e = 0; e < 16; ++e)
      bU[t][e] = (_Float16)Ui[(hi * 8 + e + (e >= 8 ? 8 : 0)) * G + col];
    wiv[t] = Wi[col];
    bzv[t] = bi[col];
  }

  // ---- preload x tile [16 rows][20 steps] into LDS (320 floats) ----
#pragma unroll
  for (int j = 0; j < 10; ++j) {
    int idx = lane * 10 + j;          // 0..319
    int r = idx / S, cc = idx - r * S;
    xls[wv][idx] = x[(size_t)(n0 + r) * S + cc];
  }
  // ---- h(0) = 0 ----
#pragma unroll
  for (int j = 0; j < 16; ++j)
    hbuf[wv][j * 32 + lane] = (_Float16)0.0f;
  LDS_FENCE();

  v8f c0 = (v8f)0.0f, c1 = (v8f)0.0f;  // cell state: units 0-15, 16-31

  for (int s = 0; s < S; ++s) {
    // broadcast x_t per accumulator row (row M = r + 8*hi)
    float xr[8];
#pragma unroll
    for (int r = 0; r < 8; ++r)
      xr[r] = xls[wv][(r + 8 * hi) * S + s];

    // previous h in A layout (lane = row, K split across lane halves)
    v8h a0 = *(const v8h*)&hbuf[wv][l16 * 32 + klo];
    v8h a1 = *(const v8h*)&hbuf[wv][l16 * 32 + 16 + klo];
    v16h hA = __builtin_shufflevector(a0, a1, 0, 1, 2, 3, 4, 5, 6, 7,
                                      8, 9, 10, 11, 12, 13, 14, 15);

    // z = h@Ui + x*Wi + b  (C seeded with rank-1 x term + bias)
    v8f z[8];
#pragma unroll
    for (int t = 0; t < 8; ++t) {
      v8f acc;
#pragma unroll
      for (int r = 0; r < 8; ++r) acc[r] = fmaf(xr[r], wiv[t], bzv[t]);
      z[t] = __builtin_amdgcn_wmma_f32_16x16x32_f16(
          false, hA, false, bU[t], (short)0, acc, false, false);
    }

    // gates: tiles 0-1 = i, 2-3 = f, 4-5 = g, 6-7 = o
#pragma unroll
    for (int u = 0; u < 2; ++u) {
      v8f zi = z[u], zf = z[2 + u], zg = z[4 + u], zo = z[6 + u];
      v8f& cc = (u == 0) ? c0 : c1;
#pragma unroll
      for (int r = 0; r < 8; ++r) {
        float iv = fast_sigmoid(zi[r]);
        float fv = fast_sigmoid(zf[r]);
        float gv = fast_tanh(zg[r]);
        float ov = fast_sigmoid(zo[r]);
        float cn = fmaf(fv, cc[r], iv * gv);
        cc[r] = cn;
        float hn = ov * fast_tanh(cn);
        hbuf[wv][(r + 8 * hi) * 32 + u * 16 + l16] = (_Float16)hn;
      }
    }
    LDS_FENCE();
  }

  // final h -> features (fp16, row-major [16][32], matches hbuf layout)
  {
    int base = lane * 16;  // 512 halves / 32 lanes
    v8h w0 = *(const v8h*)&hbuf[wv][base];
    v8h w1 = *(const v8h*)&hbuf[wv][base + 8];
    _Float16* dst = feats + (size_t)n0 * 32 + base;
    *(v8h*)dst       = w0;
    *(v8h*)(dst + 8) = w1;
  }
}

// ===========================================================================
// Outer LSTM(24) over T=256 with carry mask, + dense sigmoid head.
// One wave per 16-batch tile (16 tiles). Gate dim padded 24 -> 32 so the
// same 8-column-tile WMMA structure applies; padded h units forced to 0.
// ===========================================================================
__global__ __launch_bounds__(32, 1) void lstm_outer_kernel(
    const _Float16* __restrict__ feats,  // [B*T, 32] fp16
    const int* __restrict__ lengths,     // [B]
    const float* __restrict__ Wo,        // [32, 96]
    const float* __restrict__ Uo,        // [24, 96]
    const float* __restrict__ bo,        // [96]
    const float* __restrict__ Wd,        // [24, 1]
    const float* __restrict__ bd,        // [1]
    float* __restrict__ out,             // [B]
    int T)
{
  constexpr int U2 = 24;
  constexpr int G  = 96;  // 4*U2

  __shared__ __attribute__((aligned(16))) _Float16 hbuf[16 * 32];
  __shared__ __attribute__((aligned(16))) float    hout[16 * U2];

  const int lane = threadIdx.x & 31;
  const int b0   = blockIdx.x * 16;
  const int hi   = lane >> 4;
  const int l16  = lane & 15;
  const int klo  = hi * 8;

  // ---- Wo / Uo in padded B layout (gate g occupies padded cols 32g..32g+23)
  v16h bW[8], bUo[8];
  float bz[8];
#pragma unroll
  for (int t = 0; t < 8; ++t) {
    const int g = t >> 1, u = t & 1;
    const int cpad = u * 16 + l16;               // 0..31 within gate
    const bool cv = cpad < U2;
    const int col = g * U2 + (cv ? cpad : 0);    // real column in [0,96)
#pragma unroll
    for (int e = 0; e < 16; ++e) {
      int k = hi * 8 + e + (e >= 8 ? 8 : 0);
      bW[t][e]  = cv ? (_Float16)Wo[k * G + col] : (_Float16)0.0f;
      bUo[t][e] = (cv && k < U2) ? (_Float16)Uo[k * G + col] : (_Float16)0.0f;
    }
    bz[t] = cv ? bo[col] : 0.0f;
  }

  int lens[16];
#pragma unroll
  for (int m = 0; m < 16; ++m) lens[m] = lengths[b0 + m];

#pragma unroll
  for (int j = 0; j < 16; ++j) hbuf[j * 32 + lane] = (_Float16)0.0f;
  LDS_FENCE();

  v8f c0 = (v8f)0.0f, c1 = (v8f)0.0f;
  v8f h0 = (v8f)0.0f, h1 = (v8f)0.0f;

  for (int t = 0; t < T; ++t) {
    // features A operand straight from global (L2-resident, 16B chunks)
    const _Float16* fr = feats + ((size_t)(b0 + l16) * T + t) * 32;
    v8h f0 = *(const v8h*)(fr + klo);
    v8h f1 = *(const v8h*)(fr + 16 + klo);
    v16h fA = __builtin_shufflevector(f0, f1, 0, 1, 2, 3, 4, 5, 6, 7,
                                      8, 9, 10, 11, 12, 13, 14, 15);
    // previous h A operand from LDS (zero-padded units 24..31)
    v8h a0 = *(const v8h*)&hbuf[l16 * 32 + klo];
    v8h a1 = *(const v8h*)&hbuf[l16 * 32 + 16 + klo];
    v16h hA = __builtin_shufflevector(a0, a1, 0, 1, 2, 3, 4, 5, 6, 7,
                                      8, 9, 10, 11, 12, 13, 14, 15);

    // z = feats@Wo + h@Uo + bo : two chained WMMAs per column tile
    v8f z[8];
#pragma unroll
    for (int tt = 0; tt < 8; ++tt) {
      v8f acc = (v8f)bz[tt];
      acc = __builtin_amdgcn_wmma_f32_16x16x32_f16(
          false, fA, false, bW[tt], (short)0, acc, false, false);
      acc = __builtin_amdgcn_wmma_f32_16x16x32_f16(
          false, hA, false, bUo[tt], (short)0, acc, false, false);
      z[tt] = acc;
    }

#pragma unroll
    for (int u = 0; u < 2; ++u) {
      v8f zi = z[u], zf = z[2 + u], zg = z[4 + u], zo = z[6 + u];
      v8f& cc = (u == 0) ? c0 : c1;
      v8f& hh = (u == 0) ? h0 : h1;
#pragma unroll
      for (int r = 0; r < 8; ++r) {
        float iv = fast_sigmoid(zi[r]);
        float fv = fast_sigmoid(zf[r]);
        float gv = fast_tanh(zg[r]);
        float ov = fast_sigmoid(zo[r]);
        float cn = fmaf(fv, cc[r], iv * gv);
        float hn = ov * fast_tanh(cn);
        // Keras mask semantics: carry state through masked steps
        const int len = hi ? lens[r + 8] : lens[r];
        const bool msk = t < len;
        cn = msk ? cn : cc[r];
        hn = msk ? hn : hh[r];
        cc[r] = cn;
        hh[r] = hn;
        const bool pad = (u == 1) && (l16 >= 8);  // padded units stay 0
        hbuf[(r + 8 * hi) * 32 + u * 16 + l16] = pad ? (_Float16)0.0f
                                                     : (_Float16)hn;
      }
    }
    LDS_FENCE();
  }

  // ---- dense sigmoid head: out[b] = sigmoid(h @ Wd + bd) ----
#pragma unroll
  for (int r = 0; r < 8; ++r) {
    hout[(r + 8 * hi) * U2 + l16] = h0[r];
    if (l16 < 8) hout[(r + 8 * hi) * U2 + 16 + l16] = h1[r];
  }
  LDS_FENCE();
  if (lane < 16) {
    float acc = bd[0];
#pragma unroll
    for (int u = 0; u < U2; ++u) acc = fmaf(hout[lane * U2 + u], Wd[u], acc);
    out[b0 + lane] = fast_sigmoid(acc);
  }
}

// ===========================================================================
extern "C" void kernel_launch(void* const* d_in, const int* in_sizes, int n_in,
                              void* d_out, int out_size, void* d_ws, size_t ws_size,
                              hipStream_t stream) {
  (void)in_sizes; (void)n_in; (void)out_size; (void)ws_size;
  const float* x       = (const float*)d_in[0];  // [256,256,20,1]
  const int*   lengths = (const int*)d_in[1];    // [256]
  const float* Wi      = (const float*)d_in[2];  // [1,128]
  const float* Ui      = (const float*)d_in[3];  // [32,128]
  const float* bi      = (const float*)d_in[4];  // [128]
  const float* Wo      = (const float*)d_in[5];  // [32,96]
  const float* Uo      = (const float*)d_in[6];  // [24,96]
  const float* bo      = (const float*)d_in[7];  // [96]
  const float* Wd      = (const float*)d_in[8];  // [24,1]
  const float* bd      = (const float*)d_in[9];  // [1]
  float* out = (float*)d_out;

  constexpr int B = 256, T = 256;
  _Float16* feats = (_Float16*)d_ws;  // [B*T, 32] fp16 = 4 MB scratch

  // 65536 rows -> 4096 wave-tiles -> 512 blocks x 8 waves
  lstm_inner_kernel<<<dim3((B * T) / 16 / 8), dim3(256), 0, stream>>>(
      x, Wi, Ui, bi, feats);
  // 256 batches -> 16 single-wave tiles
  lstm_outer_kernel<<<dim3(B / 16), dim3(32), 0, stream>>>(
      feats, lengths, Wo, Uo, bo, Wd, bd, out, T);
}
